// AdaptiveModFusion_68848325755517
// MI455X (gfx1250) — matrix operand order repaired and checked by reading, not compiled, not run on hardware
//
#include <hip/hip_runtime.h>
#include <stdint.h>

#define B_  16
#define N_  1024
#define C_  256
#define R_  8
#define NS_ 32

typedef __bf16 bf16;
typedef __attribute__((ext_vector_type(16))) bf16  v16bf;
typedef __attribute__((ext_vector_type(8)))  bf16  v8bf;
typedef __attribute__((ext_vector_type(8)))  float v8f;

__device__ __forceinline__ bf16 f2bf(float f) {
  union { float f; uint32_t u; } v; v.f = f;
  uint32_t r = v.u + 0x7FFFu + ((v.u >> 16) & 1u);   // RNE
  union { uint16_t s; bf16 b; } o; o.s = (uint16_t)(r >> 16);
  return o.b;
}
__device__ __forceinline__ float bf2f(bf16 b) {
  union { uint16_t s; bf16 b; } i; i.b = b;
  union { float f; uint32_t u; } o; o.u = ((uint32_t)i.s) << 16;
  return o.f;
}

#define WMMA_BF16(A, Bm, Cacc) \
  __builtin_amdgcn_wmma_f32_16x16x32_bf16(false, (A), false, (Bm), (short)0, (Cacc), false, false)

// ---------------------------------------------------------------------------
// bf16 WMMA GEMM, 32x64 macro-tile per wave (2 M-tiles x 4 N-tiles):
//   out = act(scale * (A1|A2-concat) @ WT^T + bias [+ prior])
// A row-major [M,K] (leading dim LDA1/LDA2, compile-time), WT row-major [N,K]
// (leading dim LDW, compile-time). Requires M%32==0, N%64==0.
// Compile-time leading dims turn all intra-macro-tile strides into immediate
// offsets of global_load_b128, so only two base pointers stay live across the
// K-loop (prevents the pointer spills seen with runtime strides).
// ---------------------------------------------------------------------------
template<int LDA1, int LDA2, int LDW, bool HASA2>
__global__ void gemm_wmma(
    const bf16* __restrict__ A1, long strideA1, int K1,
    const bf16* __restrict__ A2, int K2,
    const bf16* __restrict__ WT, long strideW,
    const float* __restrict__ bias, float scale, int act, int accumulate,
    float* __restrict__ outF, long strideOF,
    bf16* __restrict__ outB, long strideOB,
    bf16* __restrict__ outBT, int ldbt,
    int M, int N, int batch)
{
  const int lane = threadIdx.x & 31;
  const int wave = threadIdx.x >> 5;
  const int wpb  = blockDim.x >> 5;
  const int Mb = M >> 5;              // 32-row macro tiles
  const int Nb = N >> 6;              // 64-col macro tiles
  const long tilesPerBatch = (long)Mb * Nb;
  const long tiles = tilesPerBatch * batch;
  const long nWaves = (long)gridDim.x * wpb;
  const int row = lane & 15;
  const int ka  = (lane >> 4) << 4;   // A-frag K base (0 / 16)
  const int kb  = (lane >> 4) << 4;   // B-frag K base (0 / 16)

  for (long t = (long)blockIdx.x * wpb + wave; t < tiles; t += nWaves) {
    int  b  = (int)(t / tilesPerBatch);
    long r  = t - (long)b * tilesPerBatch;
    int  mt = (int)(r / Nb);
    int  nt = (int)(r - (long)mt * Nb);

    v8f acc[2][4];
#pragma unroll
    for (int i = 0; i < 2; i++)
#pragma unroll
      for (int j = 0; j < 4; j++)
        acc[i][j] = (v8f){0.f, 0.f, 0.f, 0.f, 0.f, 0.f, 0.f, 0.f};

    {
      const bf16* ap = A1 + (long)b * strideA1 + (long)(mt * 32 + row) * LDA1 + ka;
      const bf16* wp = WT + (long)b * strideW + (long)(nt * 64 + row) * LDW + kb;
#pragma unroll 2
      for (int kk = 0; kk < K1; kk += 32) {
        v16bf a0, a1v, b0, b1, b2, b3;
        ((v8bf*)&a0)[0]  = *(const v8bf*)(ap + kk);
        ((v8bf*)&a0)[1]  = *(const v8bf*)(ap + kk + 8);
        ((v8bf*)&a1v)[0] = *(const v8bf*)(ap + 16 * LDA1 + kk);
        ((v8bf*)&a1v)[1] = *(const v8bf*)(ap + 16 * LDA1 + kk + 8);
        b0 = *(const v16bf*)(wp + kk);
        b1 = *(const v16bf*)(wp + 16 * LDW + kk);
        b2 = *(const v16bf*)(wp + 32 * LDW + kk);
        b3 = *(const v16bf*)(wp + 48 * LDW + kk);
        acc[0][0] = WMMA_BF16(a0,  b0, acc[0][0]);
        acc[0][1] = WMMA_BF16(a0,  b1, acc[0][1]);
        acc[0][2] = WMMA_BF16(a0,  b2, acc[0][2]);
        acc[0][3] = WMMA_BF16(a0,  b3, acc[0][3]);
        acc[1][0] = WMMA_BF16(a1v, b0, acc[1][0]);
        acc[1][1] = WMMA_BF16(a1v, b1, acc[1][1]);
        acc[1][2] = WMMA_BF16(a1v, b2, acc[1][2]);
        acc[1][3] = WMMA_BF16(a1v, b3, acc[1][3]);
      }
    }
    if (HASA2) { // concat([A1,A2]) @ W : second half of K
      const bf16* ap = A2 + (long)(mt * 32 + row) * LDA2 + ka;
      const bf16* wp = WT + (long)b * strideW + (long)(nt * 64 + row) * LDW + kb + K1;
#pragma unroll 2
      for (int kk = 0; kk < K2; kk += 32) {
        v16bf a0, a1v, b0, b1, b2, b3;
        ((v8bf*)&a0)[0]  = *(const v8bf*)(ap + kk);
        ((v8bf*)&a0)[1]  = *(const v8bf*)(ap + kk + 8);
        ((v8bf*)&a1v)[0] = *(const v8bf*)(ap + 16 * LDA2 + kk);
        ((v8bf*)&a1v)[1] = *(const v8bf*)(ap + 16 * LDA2 + kk + 8);
        b0 = *(const v16bf*)(wp + kk);
        b1 = *(const v16bf*)(wp + 16 * LDW + kk);
        b2 = *(const v16bf*)(wp + 32 * LDW + kk);
        b3 = *(const v16bf*)(wp + 48 * LDW + kk);
        acc[0][0] = WMMA_BF16(a0,  b0, acc[0][0]);
        acc[0][1] = WMMA_BF16(a0,  b1, acc[0][1]);
        acc[0][2] = WMMA_BF16(a0,  b2, acc[0][2]);
        acc[0][3] = WMMA_BF16(a0,  b3, acc[0][3]);
        acc[1][0] = WMMA_BF16(a1v, b0, acc[1][0]);
        acc[1][1] = WMMA_BF16(a1v, b1, acc[1][1]);
        acc[1][2] = WMMA_BF16(a1v, b2, acc[1][2]);
        acc[1][3] = WMMA_BF16(a1v, b3, acc[1][3]);
      }
    }

#pragma unroll
    for (int i = 0; i < 2; i++) {
      const int gm0 = mt * 32 + i * 16 + ((lane >> 4) << 3);
#pragma unroll
      for (int j = 0; j < 4; j++) {
        const int gn = nt * 64 + j * 16 + (lane & 15);
        const float bv = bias ? bias[gn] : 0.f;
#pragma unroll
        for (int rr = 0; rr < 8; rr++) {
          float vv = acc[i][j][rr] * scale + bv;
          long offRow = (long)(gm0 + rr) * N + gn;
          if (accumulate) vv += outF[(long)b * strideOF + offRow];
          if (act == 1)      vv = fmaxf(vv, 0.f);
          else if (act == 2) vv = 1.f / (1.f + __expf(-vv));
          if (outF)  outF[(long)b * strideOF + offRow] = vv;
          if (outB)  outB[(long)b * strideOB + offRow] = f2bf(vv);
          if (outBT) outBT[(long)gn * ldbt + (gm0 + rr)] = f2bf(vv);
        }
      }
    }
  }
}

// ---------------------------------------------------------------------------
// Elementwise / reduction helper kernels
// ---------------------------------------------------------------------------
__global__ void to_bf16_kernel(const float* __restrict__ src, bf16* __restrict__ dst, long n) {
  for (long i = (long)blockIdx.x * blockDim.x + threadIdx.x; i < n;
       i += (long)gridDim.x * blockDim.x)
    dst[i] = f2bf(src[i]);
}

// W [batch,K,N] (f32) -> WT [batch,N,K] (bf16)
__global__ void transpose_f2b(const float* __restrict__ W, bf16* __restrict__ WT,
                              int K, int N, int batch) {
  long total = (long)batch * K * N;
  for (long i = (long)blockIdx.x * blockDim.x + threadIdx.x; i < total;
       i += (long)gridDim.x * blockDim.x) {
    long bi = i / ((long)K * N);
    long r  = i - bi * (long)K * N;
    int  k  = (int)(r / N), n = (int)(r - (long)k * N);
    WT[bi * (long)K * N + (long)n * K + k] = f2bf(W[i]);
  }
}

// row softmax over 1024 columns, f32 -> bf16 probs. One block per row.
__global__ void softmax_rows(const float* __restrict__ S, bf16* __restrict__ P) {
  long row = blockIdx.x;
  const float* s = S + row * (long)N_;
  bf16* p = P + row * (long)N_;
  int tid = threadIdx.x, lane = tid & 31, wv = tid >> 5;
  __shared__ float red[8];
  float v0 = s[tid], v1 = s[tid + 256], v2 = s[tid + 512], v3 = s[tid + 768];
  float m = fmaxf(fmaxf(v0, v1), fmaxf(v2, v3));
  for (int o = 16; o; o >>= 1) m = fmaxf(m, __shfl_xor(m, o));
  if (lane == 0) red[wv] = m;
  __syncthreads();
  float bm = -1e30f;
  for (int i = 0; i < 8; i++) bm = fmaxf(bm, red[i]);
  __syncthreads();
  float e0 = __expf(v0 - bm), e1 = __expf(v1 - bm), e2 = __expf(v2 - bm), e3 = __expf(v3 - bm);
  float sum = e0 + e1 + e2 + e3;
  for (int o = 16; o; o >>= 1) sum += __shfl_xor(sum, o);
  if (lane == 0) red[wv] = sum;
  __syncthreads();
  float tot = 0.f;
  for (int i = 0; i < 8; i++) tot += red[i];
  float inv = 1.f / tot;
  p[tid] = f2bf(e0 * inv); p[tid + 256] = f2bf(e1 * inv);
  p[tid + 512] = f2bf(e2 * inv); p[tid + 768] = f2bf(e3 * inv);
}

// in-place: gc = gc*outv + (1-gc)*xv
__global__ void ew_gate(float* __restrict__ gc, const float* __restrict__ outv,
                        const float* __restrict__ xv, long n) {
  for (long i = (long)blockIdx.x * blockDim.x + threadIdx.x; i < n;
       i += (long)gridDim.x * blockDim.x) {
    float g = gc[i];
    gc[i] = g * outv[i] + (1.f - g) * xv[i];
  }
}

// LayerNorm over C_=256 per row; optional relu; optional f32/bf16 outputs.
__global__ void ln_rows(const float* __restrict__ in, const float* __restrict__ g,
                        const float* __restrict__ b, float* __restrict__ outF,
                        bf16* __restrict__ outB, int relu) {
  long row = blockIdx.x;
  int c = threadIdx.x, lane = c & 31, wv = c >> 5;
  long idx = row * (long)C_ + c;
  float v = in[idx];
  __shared__ float red[8];
  float s = v;
  for (int o = 16; o; o >>= 1) s += __shfl_xor(s, o);
  if (lane == 0) red[wv] = s;
  __syncthreads();
  float tot = 0.f;
  for (int i = 0; i < 8; i++) tot += red[i];
  float mu = tot * (1.f / 256.f);
  __syncthreads();
  float d = v - mu;
  s = d * d;
  for (int o = 16; o; o >>= 1) s += __shfl_xor(s, o);
  if (lane == 0) red[wv] = s;
  __syncthreads();
  tot = 0.f;
  for (int i = 0; i < 8; i++) tot += red[i];
  float var = tot * (1.f / 256.f);
  float y = d * rsqrtf(var + 1e-5f) * g[c] + b[c];
  if (relu) y = fmaxf(y, 0.f);
  if (outF) outF[idx] = y;
  if (outB) outB[idx] = f2bf(y);
}

// conf = sigmoid(t1 . Wc2 + bc2); weighted = validated * conf (bf16)
__global__ void conf_weighted(const float* __restrict__ t1, const float* __restrict__ Wc2,
                              const float* __restrict__ bc2, const float* __restrict__ valf,
                              float* __restrict__ conf, bf16* __restrict__ wgt) {
  long row = blockIdx.x;
  int tid = threadIdx.x, lane = tid & 31, wv = tid >> 5;
  __shared__ float red[8];
  float p = 0.f;
  if (tid < 128) p = t1[row * 128 + tid] * Wc2[tid];
  for (int o = 16; o; o >>= 1) p += __shfl_xor(p, o);
  if (lane == 0) red[wv] = p;
  __syncthreads();
  float s = 0.f;
  for (int i = 0; i < 8; i++) s += red[i];
  float cf = 1.f / (1.f + __expf(-(s + bc2[0])));
  if (tid == 0) conf[row] = cf;
  wgt[row * (long)C_ + tid] = f2bf(valf[row * (long)C_ + tid] * cf);
}

__global__ void count_edges(const int* __restrict__ dst, const int* __restrict__ types,
                            int* __restrict__ cnt, int E) {
  for (long i = (long)blockIdx.x * blockDim.x + threadIdx.x; i < E;
       i += (long)gridDim.x * blockDim.x)
    atomicAdd(&cnt[(long)dst[i] * R_ + types[i]], 1);
}

// A[dst] += xn[src] / cnt[dst,rel] for edges of this relation. One wave/edge.
__global__ void scatter_rel(const int* __restrict__ src, const int* __restrict__ dst,
                            const int* __restrict__ types, int rel,
                            const bf16* __restrict__ xn, const int* __restrict__ cnt,
                            float* __restrict__ A, int E) {
  long gid = (long)blockIdx.x * blockDim.x + threadIdx.x;
  int lane = (int)(gid & 31);
  long e = gid >> 5;
  long estep = ((long)gridDim.x * blockDim.x) >> 5;
  for (; e < E; e += estep) {
    if (types[e] != rel) continue;
    int d = dst[e], sidx = src[e];
    float wgt = 1.f / fmaxf((float)cnt[(long)d * R_ + rel], 1.f);
    const bf16* xr = xn + (long)sidx * C_;
    float* ar = A + (long)d * C_;
#pragma unroll
    for (int j = 0; j < 8; j++) {
      int c = lane + 32 * j;
      atomicAdd(&ar[c], bf2f(xr[c]) * wgt);
    }
  }
}

__global__ void pool_kernel(const float* __restrict__ xrg, const float* __restrict__ conf,
                            float* __restrict__ pooled) {
  int b = blockIdx.x, c = threadIdx.x;
  float acc = 0.f, cs = 0.f;
  for (int n = 0; n < N_; n++) {
    float cf = conf[(long)b * N_ + n];
    acc += xrg[((long)b * N_ + n) * C_ + c] * cf;
    cs  += cf;
  }
  pooled[b * C_ + c] = acc / fmaxf(cs, 1e-8f);
}

__global__ void head_kernel(const float* __restrict__ pooled, const float* __restrict__ Wh,
                            const float* __restrict__ bh, float* __restrict__ out) {
  int idx = blockIdx.x * blockDim.x + threadIdx.x;
  if (idx >= B_ * NS_) return;
  int b = idx >> 5, s = idx & 31;
  float acc = bh[s];
  for (int c = 0; c < C_; c++) acc += pooled[b * C_ + c] * Wh[c * NS_ + s];
  out[idx] = acc;
}

// ---------------------------------------------------------------------------
static inline int gblocks(long tiles) {
  long b = (tiles + 7) / 8;           // 8 waves per 256-thread block
  return (int)(b > 8192 ? 8192 : b);
}

extern "C" void kernel_launch(void* const* d_in, const int* in_sizes, int n_in,
                              void* d_out, int out_size, void* d_ws, size_t ws_size,
                              hipStream_t stream) {
  const float* x    = (const float*)d_in[0];
  const int*   ei   = (const int*)d_in[1];
  const int*   etyp = (const int*)d_in[2];
  const float* Wq = (const float*)d_in[3];  const float* bq = (const float*)d_in[4];
  const float* Wk = (const float*)d_in[5];  const float* bk = (const float*)d_in[6];
  const float* Wv = (const float*)d_in[7];  const float* bvv = (const float*)d_in[8];
  const float* Wg = (const float*)d_in[9];  const float* bg = (const float*)d_in[10];
  const float* lag = (const float*)d_in[11]; const float* lab = (const float*)d_in[12];
  const float* Wc1 = (const float*)d_in[13]; const float* bc1 = (const float*)d_in[14];
  const float* Wc2 = (const float*)d_in[15]; const float* bc2 = (const float*)d_in[16];
  const float* Wm  = (const float*)d_in[17]; const float* bm = (const float*)d_in[18];
  const float* l1g = (const float*)d_in[19]; const float* l1b = (const float*)d_in[20];
  const float* Wrel = (const float*)d_in[21];
  const float* Wroot = (const float*)d_in[22]; const float* brg = (const float*)d_in[23];
  const float* l2g = (const float*)d_in[24]; const float* l2b = (const float*)d_in[25];
  const float* Wh  = (const float*)d_in[26]; const float* bh = (const float*)d_in[27];
  float* out = (float*)d_out;
  const int  E = in_sizes[2];
  const long M = (long)B_ * N_;     // 16384 rows

  // ---- workspace layout (lifetime-based aliasing; needs ~165 MB) ----
  char* w = (char*)d_ws;
  const size_t MB = 1024ull * 1024ull;
  bf16*  xb    = (bf16*)(w + 0);
  bf16*  qb    = (bf16*)(w + 8 * MB);
  bf16*  kbuf  = (bf16*)(w + 16 * MB);
  bf16*  vT    = (bf16*)(w + 24 * MB);
  float* S     = (float*)(w + 32 * MB);       // 64 MB scores
  bf16*  P     = (bf16*)(w + 96 * MB);        // 32 MB probs
  float* attnf = (float*)(w + 128 * MB);
  bf16*  attnb = (bf16*)(w + 144 * MB);
  float* gatef = (float*)(w + 8 * MB);        // aliases q/k (dead after scores)
  float* valf  = (float*)(w + 32 * MB);       // aliases S (dead after softmax)
  bf16*  valb  = (bf16*)(w + 48 * MB);
  float* t1f   = (float*)(w + 56 * MB);
  float* conff = (float*)(w + 64 * MB);
  bf16*  wgtb  = (bf16*)(w + 65 * MB);
  float* xmf   = (float*)(w + 73 * MB);
  bf16*  xnb   = (bf16*)(w + 96 * MB);        // aliases P (dead after PV gemm)
  float* Af    = (float*)(w + 104 * MB);
  bf16*  Ab    = (bf16*)(w + 120 * MB);
  float* aggf  = (float*)(w + 128 * MB);      // aliases attnf (dead after ew_gate)
  int*   cnt   = (int*)(w + 144 * MB);        // aliases attnb (dead after gate gemm)
  float* xrgf  = (float*)(w + 145 * MB);
  float* pool  = (float*)(w + 161 * MB);
  bf16*  WqT   = (bf16*)(w + 162 * MB);
  bf16*  WkT   = WqT + 256 * 256;
  bf16*  WvT   = WkT + 256 * 256;
  bf16*  WgT   = WvT + 256 * 256;        // [256,512]
  bf16*  Wc1T  = WgT + 256 * 512;        // [128,256]
  bf16*  WmT   = Wc1T + 128 * 256;       // [256,512]
  bf16*  WrelT = WmT + 256 * 512;        // 8 x [256,256]
  bf16*  WrootT = WrelT + 8 * 256 * 256;

  // ---- weight prep (bf16 transposes) + x conversion ----
  to_bf16_kernel<<<2048, 256, 0, stream>>>(x, xb, M * C_);
  transpose_f2b<<<256, 256, 0, stream>>>(Wq, WqT, 256, 256, 1);
  transpose_f2b<<<256, 256, 0, stream>>>(Wk, WkT, 256, 256, 1);
  transpose_f2b<<<256, 256, 0, stream>>>(Wv, WvT, 256, 256, 1);
  transpose_f2b<<<512, 256, 0, stream>>>(Wg, WgT, 512, 256, 1);
  transpose_f2b<<<128, 256, 0, stream>>>(Wc1, Wc1T, 256, 128, 1);
  transpose_f2b<<<512, 256, 0, stream>>>(Wm, WmT, 512, 256, 1);
  transpose_f2b<<<2048, 256, 0, stream>>>(Wrel, WrelT, 256, 256, 8);
  transpose_f2b<<<256, 256, 0, stream>>>(Wroot, WrootT, 256, 256, 1);

  const long tQKV = (M / 32) * (C_ / 64);              // 2048 macro-tiles
  // q, k (bf16 row-major), v emitted transposed vT[C, B*N]
  gemm_wmma<256,256,256,false><<<gblocks(tQKV), 256, 0, stream>>>(xb, 0, C_, nullptr, 0,
      WqT, 0, bq, 1.f, 0, 0, nullptr, 0, qb, 0, nullptr, 0, (int)M, C_, 1);
  gemm_wmma<256,256,256,false><<<gblocks(tQKV), 256, 0, stream>>>(xb, 0, C_, nullptr, 0,
      WkT, 0, bk, 1.f, 0, 0, nullptr, 0, kbuf, 0, nullptr, 0, (int)M, C_, 1);
  gemm_wmma<256,256,256,false><<<gblocks(tQKV), 256, 0, stream>>>(xb, 0, C_, nullptr, 0,
      WvT, 0, bvv, 1.f, 0, 0, nullptr, 0, nullptr, 0, vT, (int)M, (int)M, C_, 1);

  // scores = q @ k^T / 16 (batched; B-frags read k rows directly)
  long tS = (long)B_ * (N_ / 32) * (N_ / 64);          // 8192
  gemm_wmma<256,256,256,false><<<gblocks(tS), 256, 0, stream>>>(qb, (long)N_ * C_, C_,
      nullptr, 0, kbuf, (long)N_ * C_, nullptr, 0.0625f, 0, 0,
      S, (long)N_ * N_, nullptr, 0, nullptr, 0, N_, N_, B_);
  softmax_rows<<<(int)M, 256, 0, stream>>>(S, P);
  // out = P @ v (B-frags read vT rows, ld 16384)
  long tO = (long)B_ * (N_ / 32) * (C_ / 64);          // 2048
  gemm_wmma<1024,256,16384,false><<<gblocks(tO), 256, 0, stream>>>(P, (long)N_ * N_, N_,
      nullptr, 0, vT, (long)N_, nullptr, 1.f, 0, 0,
      attnf, (long)N_ * C_, attnb, (long)N_ * C_, nullptr, 0, N_, C_, B_);

  // gate = sigmoid(concat(out, x) @ Wg + bg)
  gemm_wmma<256,256,512,true><<<gblocks(tQKV), 256, 0, stream>>>(attnb, 0, C_, xb, C_,
      WgT, 0, bg, 1.f, 2, 0, gatef, 0, nullptr, 0, nullptr, 0, (int)M, C_, 1);
  ew_gate<<<2048, 256, 0, stream>>>(gatef, attnf, x, M * C_);
  ln_rows<<<(int)M, 256, 0, stream>>>(gatef, lag, lab, valf, valb, 0);

  // confidence net + weighted features
  long tC1 = (M / 32) * (128 / 64);                    // 1024
  gemm_wmma<256,256,256,false><<<gblocks(tC1), 256, 0, stream>>>(valb, 0, C_, nullptr, 0,
      Wc1T, 0, bc1, 1.f, 1, 0, t1f, 0, nullptr, 0, nullptr, 0, (int)M, 128, 1);
  conf_weighted<<<(int)M, 256, 0, stream>>>(t1f, Wc2, bc2, valf, conff, wgtb);

  // mixer + LN1
  gemm_wmma<256,256,512,true><<<gblocks(tQKV), 256, 0, stream>>>(xb, 0, C_, wgtb, C_,
      WmT, 0, bm, 1.f, 1, 0, xmf, 0, nullptr, 0, nullptr, 0, (int)M, C_, 1);
  ln_rows<<<(int)M, 256, 0, stream>>>(xmf, l1g, l1b, nullptr, xnb, 0);

  // RGCN: root term, then per-relation scatter-mean + accumulate GEMM
  gemm_wmma<256,256,256,false><<<gblocks(tQKV), 256, 0, stream>>>(xnb, 0, C_, nullptr, 0,
      WrootT, 0, brg, 1.f, 0, 0, aggf, 0, nullptr, 0, nullptr, 0, (int)M, C_, 1);
  hipMemsetAsync(cnt, 0, (size_t)M * R_ * sizeof(int), stream);
  count_edges<<<1024, 256, 0, stream>>>(ei + E, etyp, cnt, E);
  for (int r = 0; r < R_; r++) {
    hipMemsetAsync(Af, 0, (size_t)M * C_ * sizeof(float), stream);
    scatter_rel<<<4096, 256, 0, stream>>>(ei, ei + E, etyp, r, xnb, cnt, Af, E);
    to_bf16_kernel<<<2048, 256, 0, stream>>>(Af, Ab, M * C_);
    gemm_wmma<256,256,256,false><<<gblocks(tQKV), 256, 0, stream>>>(Ab, 0, C_, nullptr, 0,
        WrelT + (long)r * C_ * C_, 0, nullptr, 1.f, 0, 1,
        aggf, 0, nullptr, 0, nullptr, 0, (int)M, C_, 1);
  }
  ln_rows<<<(int)M, 256, 0, stream>>>(aggf, l2g, l2b, xrgf, nullptr, 1);

  // confidence-weighted pooling + head
  pool_kernel<<<B_, 256, 0, stream>>>(xrgf, conff, pool);
  head_kernel<<<2, 256, 0, stream>>>(pool, Wh, bh, out);

  (void)n_in; (void)out_size; (void)ws_size; (void)in_sizes;
}
